// mlp_eqprop_88871463289518
// MI455X (gfx1250) — matrix operand order repaired and checked by reading, not compiled, not run on hardware
//
#include <hip/hip_runtime.h>
#include <hip/hip_bf16.h>
#include <stdint.h>

// ---------------------------------------------------------------------------
// EqProp free-phase relaxation, MI455X (gfx1250, wave32, WMMA).
// States fp32; GEMMs bf16 via v_wmma_f32_16x16x32_bf16 with fp32 accumulate;
// 64-wide K chunks staged with GLOBAL_LOAD_ASYNC_TO_LDS_B128 (saddr form,
// ASYNCcnt), double-buffered in LDS with a 2-stage unrolled pipeline.
// ---------------------------------------------------------------------------

typedef __attribute__((ext_vector_type(16))) __bf16 v16bf;
typedef __attribute__((ext_vector_type(8)))  float  v8f;

#define TILE 128   // block tile (M and N)
#define KC   64    // K chunk per stage
#define LDT2 72    // padded LDS row stride (bf16 elems) for a 64-wide K chunk
#define TSZ  (TILE * LDT2)

union Frag {
  v16bf v;
  uint4 u[2];
};

__device__ __forceinline__ unsigned short f2bf_rne(float f) {
  unsigned u = __float_as_uint(f);
  u += 0x7FFFu + ((u >> 16) & 1u);           // round-to-nearest-even
  return (unsigned short)(u >> 16);
}

__device__ __forceinline__ unsigned lds_off(const void* p) {
  // LDS aperture: low 32 bits of the generic address are the LDS byte address.
  return (unsigned)(uintptr_t)p;
}

// Fused:  acc = A0@B0 [+ A1@B1] ; s_new = clip(s + DT*(-s + rhop(s)*(acc+bias)))
// and r_new = bf16(rho(s_new)) for the next step's GEMMs.
// A sources are pre-packed bf16 activations [M x K]; B sources are bf16 packed
// column-major (Bp[n][k]) so every fragment fetch is a contiguous ds_load_b128.
__global__ __launch_bounds__(128)
void eqprop_gemm_step(const unsigned short* A0, const unsigned short* B0, int K0,
                      const unsigned short* A1, const unsigned short* B1, int K1,
                      const float* __restrict__ bias,
                      const float* __restrict__ s_old,
                      float* __restrict__ s_new,
                      unsigned short* __restrict__ r_new, int N) {
  // [buf][A=0/B=1][TILE rows x LDT2]; buffer/AB selection is a constant byte
  // offset folded into the async LDS-destination operand (no indexed moves).
  __shared__ __align__(16) unsigned short Tiles[2][2][TSZ];
  constexpr unsigned AB_STRIDE  = TSZ * 2;       // bytes: A tile -> B tile
  constexpr unsigned BUF_STRIDE = 2 * TSZ * 2;   // bytes: buf0 -> buf1

  const int tid  = threadIdx.x;
  const int lane = tid & 31;
  const int wave = tid >> 5;     // 4 waves
  const int wm   = wave >> 1;    // 0..1 -> 64-row slice
  const int wn   = wave & 1;     // 0..1 -> 64-col slice
  const int half = lane >> 4;
  const int l15  = lane & 15;

  const int n0 = blockIdx.x * TILE;
  const int m0 = blockIdx.y * TILE;

  // --- per-thread async slot geometry: 128 rows x 8 16B chunks = 8 slots ---
  int rowi[8], hi[8];
  unsigned so[8];                 // LDS byte offset of slot in buf0/A
#pragma unroll
  for (int i = 0; i < 8; ++i) {
    const int f = i * 128 + tid;  // 0..1023
    rowi[i] = f >> 3;
    hi[i]   = f & 7;
    so[i]   = lds_off(&Tiles[0][0][rowi[i] * LDT2 + hi[i] * 8]);
  }

  v8f acc[4][4];
#pragma unroll
  for (int i = 0; i < 4; ++i)
#pragma unroll
    for (int j = 0; j < 4; ++j)
      acc[i][j] = (v8f){0.f, 0.f, 0.f, 0.f, 0.f, 0.f, 0.f, 0.f};

  const int nk0 = K0 >> 6;        // K multiples of 64; nk always even here
  const int nk1 = K1 >> 6;
  const int nk  = nk0 + nk1;

  // Issue-state: bases/offsets for the next chunk to issue.
  const unsigned short* bA = A0 + (size_t)m0 * K0;
  const unsigned short* bB = B0 + (size_t)n0 * K0;
  unsigned voff[8];
#pragma unroll
  for (int i = 0; i < 8; ++i)
    voff[i] = (unsigned)((rowi[i] * K0 + hi[i] * 8) * 2);

  int iss = 0;  // chunk index about to be issued
  auto advance = [&]() {
    ++iss;
    if (iss == nk0 && nk1 != 0) {  // switch to second source pair
      bA = A1 + (size_t)m0 * K1;
      bB = B1 + (size_t)n0 * K1;
#pragma unroll
      for (int i = 0; i < 8; ++i)
        voff[i] = (unsigned)((rowi[i] * K1 + hi[i] * 8) * 2);
    } else {
      bA += KC;                    // +128 bytes along K
      bB += KC;
    }
  };

  // 16 async b128 global->LDS copies (8 A + 8 B); bufbyte is a compile-time
  // constant at every call site, so the destination operand is so[i] + imm.
  auto issue16 = [&](unsigned bufbyte) {
#pragma unroll
    for (int i = 0; i < 8; ++i)
      asm volatile("global_load_async_to_lds_b128 %0, %1, %2"
                   :: "v"(so[i] + bufbyte), "v"(voff[i]), "s"(bA) : "memory");
#pragma unroll
    for (int i = 0; i < 8; ++i)
      asm volatile("global_load_async_to_lds_b128 %0, %1, %2"
                   :: "v"(so[i] + bufbyte + AB_STRIDE), "v"(voff[i]), "s"(bB) : "memory");
  };

  auto compute = [&](int buf) {   // buf is a literal at every call site
#pragma unroll
    for (int ks = 0; ks < KC; ks += 32) {
      Frag a[4], b[4];
#pragma unroll
      for (int tm = 0; tm < 4; ++tm) {
        const int r = (wm * 64 + tm * 16 + l15) * LDT2 + ks;
        // A 16x32 bf16: lanes 0-15 K {0-7,16-23}; lanes 16-31 K {8-15,24-31}
        a[tm].u[0] = *(const uint4*)&Tiles[buf][0][r + half * 8];
        a[tm].u[1] = *(const uint4*)&Tiles[buf][0][r + 16 + half * 8];
      }
#pragma unroll
      for (int tn = 0; tn < 4; ++tn) {
        const int c = (wn * 64 + tn * 16 + l15) * LDT2 + ks;
        // B 32x16 bf16: lanes 0-15 col n K {0-7,16-23}; lanes 16-31 K {8-15,24-31}
        b[tn].u[0] = *(const uint4*)&Tiles[buf][1][c + half * 8];
        b[tn].u[1] = *(const uint4*)&Tiles[buf][1][c + 16 + half * 8];
      }
#pragma unroll
      for (int tm = 0; tm < 4; ++tm)
#pragma unroll
        for (int tn = 0; tn < 4; ++tn)
          acc[tm][tn] = __builtin_amdgcn_wmma_f32_16x16x32_bf16(
              false, a[tm].v, false, b[tn].v, (short)0, acc[tm][tn], false, false);
    }
  };

  // Prologue: stage chunks 0 and 1 (32 asyncs outstanding max, cnt is 6-bit).
  issue16(0);          advance();
  issue16(BUF_STRIDE); advance();

  // Steady state, 2-stage: wait for the older 16-op batch (in-order ASYNCcnt),
  // compute it, then refill that buffer with the next chunk.
  for (int c = 0; c + 2 < nk; c += 2) {
    asm volatile("s_wait_asynccnt 16" ::: "memory");
    __syncthreads();
    compute(0);
    __syncthreads();
    issue16(0);          advance();
    asm volatile("s_wait_asynccnt 16" ::: "memory");
    __syncthreads();
    compute(1);
    __syncthreads();
    issue16(BUF_STRIDE); advance();
  }
  // Drain last two chunks (nk is even).
  asm volatile("s_wait_asynccnt 16" ::: "memory");
  __syncthreads();
  compute(0);
  asm volatile("s_wait_asynccnt 0" ::: "memory");
  __syncthreads();
  compute(1);

  // Epilogue: C layout -> VGPR j holds row (j + 8*half), col l15.
#pragma unroll
  for (int tm = 0; tm < 4; ++tm) {
#pragma unroll
    for (int tn = 0; tn < 4; ++tn) {
      const int col = n0 + wn * 64 + tn * 16 + l15;
      const float bb = bias[col];
#pragma unroll
      for (int j = 0; j < 8; ++j) {
        const int row = m0 + wm * 64 + tm * 16 + half * 8 + j;
        const size_t idx = (size_t)row * N + col;
        float s   = s_old[idx];
        float g   = (s >= 0.f && s <= 1.f) ? 1.f : 0.f;   // rhop(s)
        float pre = acc[tm][tn][j] + bb;
        float ns  = s + 0.2f * (g * pre - s);             // s + DT*dsdt
        ns = fminf(fmaxf(ns, 0.f), 1.f);
        s_new[idx] = ns;
        r_new[idx] = f2bf_rne(ns);                        // rho(ns) == ns
      }
    }
  }
}

// fp32 -> bf16 (optionally with rho/clip01 fused), straight layout.
__global__ void cvt_bf16_kernel(unsigned short* __restrict__ dst,
                                const float* __restrict__ src, int n, int do_rho) {
  int i = blockIdx.x * 256 + threadIdx.x;
  if (i < n) {
    float x = src[i];
    if (do_rho) x = fminf(fmaxf(x, 0.f), 1.f);
    dst[i] = f2bf_rne(x);
  }
}

// dst[c*R + r] = bf16(src[r*C + c])  (pack transposed weight, coalesced reads)
__global__ void transpose_bf16_kernel(unsigned short* __restrict__ dst,
                                      const float* __restrict__ src, int R, int C) {
  int i = blockIdx.x * 256 + threadIdx.x;
  if (i < R * C) {
    int r = i / C, c = i - r * C;
    dst[(size_t)c * R + r] = f2bf_rne(src[i]);
  }
}

extern "C" void kernel_launch(void* const* d_in, const int* in_sizes, int n_in,
                              void* d_out, int out_size, void* d_ws, size_t ws_size,
                              hipStream_t stream) {
  (void)in_sizes; (void)n_in; (void)out_size; (void)ws_size;

  const float* s0   = (const float*)d_in[0];
  const float* s1   = (const float*)d_in[1];
  const float* s2   = (const float*)d_in[2];
  const float* data = (const float*)d_in[3];
  const float* W0   = (const float*)d_in[4];
  const float* b0   = (const float*)d_in[5];
  const float* W1   = (const float*)d_in[6];
  const float* b1   = (const float*)d_in[7];
  const float* W2   = (const float*)d_in[8];
  const float* b2   = (const float*)d_in[9];

  const size_t E0  = (size_t)2048 * 1024;   // s0 elems
  const size_t E1  = (size_t)2048 * 2048;   // s1/s2/data elems
  const size_t EW0 = (size_t)1024 * 2048;   // W0 elems

  // d_out doubles as the fp32 state "X" buffer (layout == return order).
  float* x0 = (float*)d_out;
  float* x1 = x0 + E0;
  float* x2 = x1 + E1;

  // Workspace (~120 MB): fp32 state pong, 2x bf16 activation sets, packed weights.
  uint8_t* w = (uint8_t*)d_ws;
  float* y0 = (float*)w;                     w += E0 * 4;
  float* y1 = (float*)w;                     w += E1 * 4;
  float* y2 = (float*)w;                     w += E1 * 4;
  unsigned short* rA0 = (unsigned short*)w;  w += E0 * 2;   // rho(X) bf16
  unsigned short* rA1 = (unsigned short*)w;  w += E1 * 2;
  unsigned short* rA2 = (unsigned short*)w;  w += E1 * 2;
  unsigned short* rB0 = (unsigned short*)w;  w += E0 * 2;   // rho(Y) bf16
  unsigned short* rB1 = (unsigned short*)w;  w += E1 * 2;
  unsigned short* rB2 = (unsigned short*)w;  w += E1 * 2;
  unsigned short* W0p  = (unsigned short*)w; w += EW0 * 2;  // B for r1@W0^T
  unsigned short* W0tp = (unsigned short*)w; w += EW0 * 2;  // B for r0@W0
  unsigned short* W1p  = (unsigned short*)w; w += E1 * 2;   // B for r2@W1^T
  unsigned short* W1tp = (unsigned short*)w; w += E1 * 2;   // B for r1@W1
  unsigned short* W2p  = (unsigned short*)w; w += E1 * 2;   // B for rd@W2^T
  unsigned short* rdbf = (unsigned short*)w; w += E1 * 2;   // bf16(rho(data))

  const dim3 blk256(256);

  // --- once per launch: pack weights, rho(data), rho(initial states) ---
  cvt_bf16_kernel<<<dim3((EW0 + 255) / 256), blk256, 0, stream>>>(W0p, W0, (int)EW0, 0);
  transpose_bf16_kernel<<<dim3((EW0 + 255) / 256), blk256, 0, stream>>>(W0tp, W0, 1024, 2048);
  cvt_bf16_kernel<<<dim3((E1 + 255) / 256), blk256, 0, stream>>>(W1p, W1, (int)E1, 0);
  transpose_bf16_kernel<<<dim3((E1 + 255) / 256), blk256, 0, stream>>>(W1tp, W1, 2048, 2048);
  cvt_bf16_kernel<<<dim3((E1 + 255) / 256), blk256, 0, stream>>>(W2p, W2, (int)E1, 0);
  cvt_bf16_kernel<<<dim3((E1 + 255) / 256), blk256, 0, stream>>>(rdbf, data, (int)E1, 1);
  cvt_bf16_kernel<<<dim3((E0 + 255) / 256), blk256, 0, stream>>>(rA0, s0, (int)E0, 1);
  cvt_bf16_kernel<<<dim3((E1 + 255) / 256), blk256, 0, stream>>>(rA1, s1, (int)E1, 1);
  cvt_bf16_kernel<<<dim3((E1 + 255) / 256), blk256, 0, stream>>>(rA2, s2, (int)E1, 1);

  // initial fp32 state -> X (d_out)
  hipMemcpyAsync(x0, s0, E0 * 4, hipMemcpyDeviceToDevice, stream);
  hipMemcpyAsync(x1, s1, E1 * 4, hipMemcpyDeviceToDevice, stream);
  hipMemcpyAsync(x2, s2, E1 * 4, hipMemcpyDeviceToDevice, stream);

  const dim3 blk(128);
  const dim3 g0(1024 / TILE, 2048 / TILE);   // layer 0: N=1024
  const dim3 g1(2048 / TILE, 2048 / TILE);   // layers 1,2: N=2048

  // T=30 (even): odd steps X->Y, even steps Y->X; final state lands in d_out.
  for (int t = 1; t <= 30; ++t) {
    const float *c0, *c1, *c2;
    const unsigned short *q0, *q1, *q2;     // rho(current) bf16
    float *n0p, *n1p, *n2p;
    unsigned short *o0, *o1, *o2;           // rho(next) bf16
    if (t & 1) {
      c0 = x0; c1 = x1; c2 = x2; q0 = rA0; q1 = rA1; q2 = rA2;
      n0p = y0; n1p = y1; n2p = y2; o0 = rB0; o1 = rB1; o2 = rB2;
    } else {
      c0 = y0; c1 = y1; c2 = y2; q0 = rB0; q1 = rB1; q2 = rB2;
      n0p = x0; n1p = x1; n2p = x2; o0 = rA0; o1 = rA1; o2 = rA2;
    }

    // d0 = -s0 + rhop(s0)*(rho(s1)@W0^T + b0)
    eqprop_gemm_step<<<g0, blk, 0, stream>>>(
        q1, W0p, 2048, nullptr, nullptr, 0, b0, c0, n0p, o0, 1024);
    // d1 = -s1 + rhop(s1)*(rho(s2)@W1^T + b1 + rho(s0)@W0)
    eqprop_gemm_step<<<g1, blk, 0, stream>>>(
        q2, W1p, 2048, q0, W0tp, 1024, b1, c1, n1p, o1, 2048);
    // d2 = -s2 + rhop(s2)*(rho(data)@W2^T + b2 + rho(s1)@W1)
    eqprop_gemm_step<<<g1, blk, 0, stream>>>(
        rdbf, W2p, 2048, q1, W1tp, 2048, b2, c2, n2p, o2, 2048);
  }
}
